// ACAN_78082505441950
// MI455X (gfx1250) — compile-verified, matrix operand order checked by reading, not run on hardware
//
#include <hip/hip_runtime.h>
#include <hip/hip_bf16.h>

typedef __attribute__((ext_vector_type(16))) __bf16 v16bf;
typedef __attribute__((ext_vector_type(8)))  float  v8f;

#define C_IN 256
#define D_K  128
#define HW   16384
#define LENK 18
#define TDIM 256
#define EPS  1e-5f

// d_ws layout (float offsets)
#define FRAG_OFF 0            // bf16 B-fragments: Wq 16384 floats, Wk 16384 floats (64KB each)
#define TK_OFF   32768        // 8*18*128
#define MU_OFF   51200        // 8*18*256
#define SG_OFF   88064        // 8*18*256
#define SUM_OFF  124928       // 8*256
#define SQ_OFF   126976       // 8*256
// total 129024 floats = ~504 KB

__global__ __launch_bounds__(256) void acan_zero(float* __restrict__ p, int n) {
  int i = blockIdx.x * blockDim.x + threadIdx.x;
  if (i < n) p[i] = 0.f;
}

// Per-(n,c) sums: grid = 8 n * 32 chunks, 256 threads = channel lane (coalesced)
__global__ __launch_bounds__(256) void acan_stats(const float* __restrict__ x,
                                                  float* __restrict__ sums,
                                                  float* __restrict__ sqs) {
  int n = blockIdx.x >> 5;
  int chunk = blockIdx.x & 31;
  int c = threadIdx.x;
  const float* base = x + (((size_t)n * HW + (size_t)chunk * 512) * C_IN) + c;
  float s = 0.f, q = 0.f;
  for (int p = 0; p < 512; ++p) {
    float v = base[(size_t)p * C_IN];
    s += v; q += v * v;
  }
  atomicAdd(&sums[n * C_IN + c], s);
  atomicAdd(&sqs[n * C_IN + c], q);
}

// t_k = t@Wtk+btk, mu_t = t@Wmu+bmu, sigma_t = t@Wsig+bsig ; one block per (n,key)
__global__ __launch_bounds__(256) void acan_tproj(
    const float* __restrict__ t,
    const float* __restrict__ Wtk, const float* __restrict__ btk,
    const float* __restrict__ Wmu, const float* __restrict__ bmu,
    const float* __restrict__ Wsg, const float* __restrict__ bsg,
    float* __restrict__ ws) {
  __shared__ float trow[TDIM];
  const int nk = blockIdx.x;           // n*LENK + k
  const int j = threadIdx.x;
  trow[j] = t[(size_t)nk * TDIM + j];
  __syncthreads();
  if (j < D_K) {
    float s = btk[j];
    for (int i = 0; i < TDIM; ++i) s += trow[i] * Wtk[i * D_K + j];
    ws[TK_OFF + nk * D_K + j] = s;
  }
  float sm = bmu[j], ss = bsg[j];
  for (int i = 0; i < TDIM; ++i) {
    float tv = trow[i];
    sm += tv * Wmu[i * C_IN + j];
    ss += tv * Wsg[i * C_IN + j];
  }
  ws[MU_OFF + nk * C_IN + j] = sm;
  ws[SG_OFF + nk * C_IN + j] = ss;
}

// Pack Wq/Wk (256x128 f32, row-major) into CDNA5 bf16 B-fragment layout.
// Fragment (kt,nt) covers K = kt*32..+31, N = nt*16..+15.
// Lane l<16: col=l,   kbase=0 ; lane>=16: col=l-16, kbase=8.
// Element e -> K = kbase + e + (e>=8 ? 8 : 0)   (matches 16-bit A/B striping)
__global__ __launch_bounds__(256) void acan_packw(const float* __restrict__ Wq,
                                                  const float* __restrict__ Wk,
                                                  float* __restrict__ ws) {
  int id = blockIdx.x * 256 + threadIdx.x;   // 0..4095
  int lane = id & 31;
  int nt = (id >> 5) & 7;
  int kt = (id >> 8) & 7;
  int m  = (id >> 11) & 1;
  const float* W = m ? Wk : Wq;
  int col = nt * 16 + (lane & 15);
  int kb  = kt * 32 + ((lane >> 4) << 3);
  v16bf v;
  #pragma unroll
  for (int e = 0; e < 16; ++e) {
    int K = kb + e + ((e >= 8) ? 8 : 0);
    v[e] = (__bf16)W[K * D_K + col];
  }
  v16bf* dst = (v16bf*)(ws + FRAG_OFF) + ((size_t)m * 2048 + (kt * 8 + nt) * 32 + lane);
  *dst = v;
}

// Fused main kernel: 4 waves/block, each wave = 2 M-tiles = 32 pixels.
// Every B-fragment load is shared by 4 WMMAs (Q/K x 2 tiles) -> halved L2 weight traffic.
__global__ __launch_bounds__(128) void acan_main(
    const float* __restrict__ x, const float* __restrict__ bq, const float* __restrict__ bk,
    const float* __restrict__ ws, float* __restrict__ out) {
  __shared__ float tkS[LENK][D_K + 1];     // +1 pad: conflict-free
  __shared__ float muS[LENK][C_IN];
  __shared__ float sgS[LENK][C_IN];
  __shared__ float meanS[C_IN], istdS[C_IN];
  __shared__ float xqS[4][32][D_K + 1];    // per-wave x_q (32 pixels), padded
  __shared__ float stS[4][32][20];         // [0..17]=st, [18]=sx(soft), [19]=sx(raw)

  const int tid = threadIdx.x;
  const int n = blockIdx.x >> 7;           // 128 blocks per batch image (128 px/block)

  for (int i = tid; i < LENK * D_K; i += 128)
    tkS[i / D_K][i % D_K] = ws[TK_OFF + n * LENK * D_K + i];
  for (int i = tid; i < LENK * C_IN; i += 128) {
    muS[i / C_IN][i % C_IN] = ws[MU_OFF + n * LENK * C_IN + i];
    sgS[i / C_IN][i % C_IN] = ws[SG_OFF + n * LENK * C_IN + i];
  }
  for (int c = tid; c < C_IN; c += 128) {
    float s = ws[SUM_OFF + n * C_IN + c];
    float q = ws[SQ_OFF  + n * C_IN + c];
    float m = s * (1.f / HW);
    float v = fmaxf(q * (1.f / HW) - m * m, 0.f);
    meanS[c] = m;
    istdS[c] = rsqrtf(v + EPS);
  }
  __syncthreads();

  const int wave = tid >> 5;
  const int lane = tid & 31;
  const int l16  = lane & 15;
  const int half = lane >> 4;

  const size_t gpix = (size_t)blockIdx.x * 128 + wave * 32;
  const float* xrow0 = x + (gpix + l16) * C_IN;
  const float* xrow1 = x + (gpix + 16 + l16) * C_IN;

  // ---- Phase 1: A fragments for both M-tiles (bf16, WMMA striping)
  v16bf a0[8], a1[8];
  #pragma unroll
  for (int kt = 0; kt < 8; ++kt) {
    const float* p0 = xrow0 + kt * 32 + half * 8;
    const float* p1 = xrow1 + kt * 32 + half * 8;
    #pragma unroll
    for (int e = 0; e < 8; ++e) {
      a0[kt][e]     = (__bf16)p0[e];
      a0[kt][e + 8] = (__bf16)p0[e + 16];
      a1[kt][e]     = (__bf16)p1[e];
      a1[kt][e + 8] = (__bf16)p1[e + 16];
    }
  }

  // ---- Phase 2: Q/K projections (WMMA), scores_x partials, spill x_q to LDS
  float sxp0[8], sxp1[8];
  #pragma unroll
  for (int r = 0; r < 8; ++r) { sxp0[r] = 0.f; sxp1[r] = 0.f; }

  const v16bf* fragQ = (const v16bf*)(ws + FRAG_OFF);
  const v16bf* fragK = fragQ + 2048;

  for (int nt = 0; nt < 8; ++nt) {
    v8f q0 = {0.f,0.f,0.f,0.f,0.f,0.f,0.f,0.f};
    v8f k0 = {0.f,0.f,0.f,0.f,0.f,0.f,0.f,0.f};
    v8f q1 = {0.f,0.f,0.f,0.f,0.f,0.f,0.f,0.f};
    v8f k1 = {0.f,0.f,0.f,0.f,0.f,0.f,0.f,0.f};
    #pragma unroll
    for (int kt = 0; kt < 8; ++kt) {
      v16bf bqf = fragQ[(kt * 8 + nt) * 32 + lane];
      v16bf bkf = fragK[(kt * 8 + nt) * 32 + lane];
      q0 = __builtin_amdgcn_wmma_f32_16x16x32_bf16(false, a0[kt], false, bqf, (short)0, q0, false, false);
      k0 = __builtin_amdgcn_wmma_f32_16x16x32_bf16(false, a0[kt], false, bkf, (short)0, k0, false, false);
      q1 = __builtin_amdgcn_wmma_f32_16x16x32_bf16(false, a1[kt], false, bqf, (short)0, q1, false, false);
      k1 = __builtin_amdgcn_wmma_f32_16x16x32_bf16(false, a1[kt], false, bkf, (short)0, k1, false, false);
    }
    float bqv = bq[nt * 16 + l16];     // C/D layout: lane holds column N = lane&15
    float bkv = bk[nt * 16 + l16];
    #pragma unroll
    for (int r = 0; r < 8; ++r) {
      float qv0 = q0[r] + bqv, kv0 = k0[r] + bkv;
      float qv1 = q1[r] + bqv, kv1 = k1[r] + bkv;
      sxp0[r] += qv0 * kv0;
      sxp1[r] += qv1 * kv1;
      xqS[wave][r + half * 8][nt * 16 + l16]      = qv0;   // tile0 row M = r + 8*half
      xqS[wave][16 + r + half * 8][nt * 16 + l16] = qv1;   // tile1
    }
  }

  // scores_x: reduce across the 16 columns held by each 16-lane half
  #pragma unroll
  for (int mask = 1; mask < 16; mask <<= 1) {
    #pragma unroll
    for (int r = 0; r < 8; ++r) {
      sxp0[r] += __shfl_xor(sxp0[r], mask, 16);
      sxp1[r] += __shfl_xor(sxp1[r], mask, 16);
    }
  }
  if (l16 == 0) {
    #pragma unroll
    for (int r = 0; r < 8; ++r) {
      stS[wave][r + half * 8][19]      = sxp0[r];
      stS[wave][16 + r + half * 8][19] = sxp1[r];
    }
  }
  __syncthreads();

  // ---- Phase 3: per-pixel scores_t + stable softmax (lane = pixel, all 32 lanes)
  {
    float sc[19];
    sc[18] = stS[wave][lane][19];
    float mx = sc[18];
    for (int k = 0; k < LENK; ++k) {
      float s = 0.f;
      #pragma unroll 4
      for (int d = 0; d < D_K; ++d)
        s += xqS[wave][lane][d] * tkS[k][d];
      sc[k] = s;
      mx = fmaxf(mx, s);
    }
    float sum = 0.f;
    #pragma unroll
    for (int k = 0; k < 19; ++k) { sc[k] = __expf(sc[k] - mx); sum += sc[k]; }
    float inv = 1.f / sum;
    #pragma unroll
    for (int k = 0; k < 19; ++k) stS[wave][lane][k] = sc[k] * inv;
  }
  __syncthreads();

  // ---- Phase 4: epilogue, all 32 lanes across channels, 32 pixels
  const float* xb = x + gpix * C_IN;
  float* ob = out + gpix * C_IN;
  for (int p = 0; p < 32; ++p) {
    float st[LENK];
    #pragma unroll
    for (int k = 0; k < LENK; ++k) st[k] = stS[wave][p][k];
    float sx = stS[wave][p][18];
    #pragma unroll
    for (int ci = 0; ci < 8; ++ci) {
      int c = lane + ci * 32;
      float wmu = 0.f, wsg = 0.f;
      #pragma unroll
      for (int k = 0; k < LENK; ++k) {
        wmu += st[k] * muS[k][c];
        wsg += st[k] * sgS[k][c];
      }
      float xv = xb[(size_t)p * C_IN + c];
      float xn = (xv - meanS[c]) * istdS[c];
      ob[(size_t)p * C_IN + c] = (1.f + wsg) * xn + wmu + sx * xv;
    }
  }
}

extern "C" void kernel_launch(void* const* d_in, const int* in_sizes, int n_in,
                              void* d_out, int out_size, void* d_ws, size_t ws_size,
                              hipStream_t stream) {
  const float* x   = (const float*)d_in[0];
  const float* t   = (const float*)d_in[1];
  const float* Wq  = (const float*)d_in[2];
  const float* bq  = (const float*)d_in[3];
  const float* Wk  = (const float*)d_in[4];
  const float* bk  = (const float*)d_in[5];
  const float* Wtk = (const float*)d_in[6];
  const float* btk = (const float*)d_in[7];
  const float* Wmu = (const float*)d_in[8];
  const float* bmu = (const float*)d_in[9];
  const float* Wsg = (const float*)d_in[10];
  const float* bsg = (const float*)d_in[11];
  float* out = (float*)d_out;
  float* ws  = (float*)d_ws;

  acan_zero <<<16,  256, 0, stream>>>(ws + SUM_OFF, 4096);
  acan_stats<<<256, 256, 0, stream>>>(x, ws + SUM_OFF, ws + SQ_OFF);
  acan_tproj<<<144, 256, 0, stream>>>(t, Wtk, btk, Wmu, bmu, Wsg, bsg, ws);
  acan_packw<<<16,  256, 0, stream>>>(Wq, Wk, ws);
  acan_main <<<1024,128, 0, stream>>>(x, bq, bk, ws, out);
}
